// Mamba_CNN_1_11897059410539
// MI455X (gfx1250) — compile-verified, matrix operand order checked by reading, not run on hardware
//
#include <hip/hip_runtime.h>
#include <hip/hip_bf16.h>
#include <math.h>

typedef __attribute__((ext_vector_type(16))) _Float16 v16h;
typedef __attribute__((ext_vector_type(8)))  float    v8f;

namespace {
constexpr int B_  = 2;
constexpr int C_  = 96;
constexpr int H_  = 64;
constexpr int W_  = 64;
constexpr int HW_ = H_ * W_;      // 4096
constexpr int DI_ = 192;
constexpr int N_  = 16;           // state dim
constexpr int R_  = 6;
constexpr int K_  = 4;            // scan directions
constexpr int HD_ = 8;            // heads
constexpr int CH_ = 12;           // C_/HD_
}

// ---------------------------------------------------------------------------
// Generic WMMA GEMM:  Out[M,N] = act( A[M,K] * B[K,N] + bias + res )
// A modes: 0 row-major (lda), 1 im2col 3x3 SAME from NCHW (Cin), 2 1x1 NCHW (Cin)
// B modes: 0 weight [N][K] row-major, 1 row-major [K][N] (ldb), 2 scan-dir gather from xc
// out modes: 0 row-major [M][N], 1 NCHW (b = m>>12, l = m&4095)
//
// LDS fragments are stored pre-permuted so each lane's 16-half WMMA fragment
// is one aligned 32-byte contiguous read (2x ds_load_b128), not 32 scalar picks.
// Fragment K-order: lanes 0-15 need K=[0..7,16..23], lanes 16-31 K=[8..15,24..31]
// -> store K=c at position p = (c&7) | ((c&8)<<1) | ((c&16)>>1).
// ---------------------------------------------------------------------------
__device__ __forceinline__ int frag_perm(int c) {
  return (c & 7) | ((c & 8) << 1) | ((c & 16) >> 1);
}

__device__ __forceinline__ float gemm_loadA(const float* __restrict__ A, int m, int k,
                                            int M, int K, int amode, int lda, int Cin) {
  if (m >= M || k >= K) return 0.f;
  if (amode == 0) return A[(long)m * lda + k];
  int b = m >> 12;
  int l = m & (HW_ - 1);
  if (amode == 2) return A[((long)(b * Cin + k)) * HW_ + l];
  // amode 1: 3x3 SAME
  int y = l >> 6, x = l & 63;
  int c = k / 9, t = k - c * 9;
  int yy = y + (t / 3) - 1;
  int xx = x + (t % 3) - 1;
  if ((unsigned)yy >= (unsigned)H_ || (unsigned)xx >= (unsigned)W_) return 0.f;
  return A[((long)(b * Cin + c)) * HW_ + yy * W_ + xx];
}

__device__ __forceinline__ int scan_gather_idx(int dir, int l) {
  if (dir == 0) return l;
  if (dir == 1) return ((l & 63) << 6) + (l >> 6);
  if (dir == 2) return HW_ - 1 - l;
  int l2 = HW_ - 1 - l;
  return ((l2 & 63) << 6) + (l2 >> 6);
}

__device__ __forceinline__ float gemm_loadB(const float* __restrict__ Bp, int k, int n,
                                            int K, int N, int bmode, int ldb, int dir) {
  if (k >= K || n >= N) return 0.f;
  if (bmode == 0) return Bp[(long)n * K + k];
  if (bmode == 1) return Bp[(long)k * ldb + n];
  // bmode 2: Bp = xc + b*DI*HW; row k = channel d, col n = l (direction gather)
  return Bp[(long)k * HW_ + scan_gather_idx(dir, n)];
}

__global__ void wmma_gemm(const float* __restrict__ Ap, const float* __restrict__ Bp,
                          const float* __restrict__ bias, const float* __restrict__ res,
                          float* __restrict__ Out,
                          int M, int N, int K,
                          int amode, int lda, int Cin,
                          int bmode, int ldb, int dir,
                          int act, int outmode,
                          long sAz, long sBz, long sOz) {
  const float* Az = Ap + sAz * blockIdx.z;
  const float* Bz = Bp + sBz * blockIdx.z;
  // row stride 48 halfs = 96 B (multiple of 32) -> every fragment read is
  // a 32B-aligned v16h load.
  __shared__ _Float16 Afrag[64][48] __attribute__((aligned(32)));
  __shared__ _Float16 Bfrag[32][48] __attribute__((aligned(32)));
  const int tid  = threadIdx.x;      // 0..127 (4 waves)
  const int wave = tid >> 5;
  const int lane = tid & 31;
  const int half = lane >> 4;
  const int mn   = lane & 15;
  const int tileM = blockIdx.x * 64;
  const int tileN = blockIdx.y * 32;
  v8f acc0 = {};
  v8f acc1 = {};
  for (int k0 = 0; k0 < K; k0 += 32) {
    // stage A: 64 rows x 32 k (pre-permuted positions)
    for (int i = tid; i < 64 * 32; i += 128) {
      int r = i >> 5, c = i & 31;
      Afrag[r][frag_perm(c)] =
          (_Float16)gemm_loadA(Az, tileM + r, k0 + c, M, K, amode, lda, Cin);
    }
    // stage B: 32 cols x 32 k (pre-permuted positions)
    for (int i = tid; i < 32 * 32; i += 128) {
      int col = i >> 5, c = i & 31;
      Bfrag[col][frag_perm(c)] =
          (_Float16)gemm_loadB(Bz, k0 + c, tileN + col, K, N, bmode, ldb, dir);
    }
    __syncthreads();
    const v16h a  = *(const v16h*)&Afrag[wave * 16 + mn][half * 16];
    const v16h b0 = *(const v16h*)&Bfrag[mn][half * 16];
    const v16h b1 = *(const v16h*)&Bfrag[16 + mn][half * 16];
    acc0 = __builtin_amdgcn_wmma_f32_16x16x32_f16(false, a, false, b0, (short)0, acc0,
                                                  false, false);
    acc1 = __builtin_amdgcn_wmma_f32_16x16x32_f16(false, a, false, b1, (short)0, acc1,
                                                  false, false);
    __syncthreads();
  }
#pragma unroll
  for (int g = 0; g < 2; ++g) {
    const int n = tileN + g * 16 + mn;
    if (n >= N) continue;
    const v8f acc = g ? acc1 : acc0;
#pragma unroll
    for (int r = 0; r < 8; ++r) {
      int m = tileM + wave * 16 + half * 8 + r;
      if (m >= M) continue;
      float v = acc[r];
      if (bias) v += bias[n];
      long oidx;
      if (outmode == 0) oidx = (long)m * N + n;
      else { int b = m >> 12; int l = m & (HW_ - 1); oidx = ((long)(b * N + n)) * HW_ + l; }
      oidx += sOz * blockIdx.z;
      if (res) v += res[oidx];
      if (act == 1) v = fmaxf(v, 0.f);
      Out[oidx] = v;
    }
  }
}

// ---------------------------------------------------------------------------
// LayerNorm over C=96 of NCHW input -> (B*L, 96) row-major
// ---------------------------------------------------------------------------
__global__ void ln_nchw_to_nlc(const float* __restrict__ in, const float* __restrict__ g,
                               const float* __restrict__ bb, float* __restrict__ out) {
  __shared__ float s[128];
  int bl = blockIdx.x;                  // b*HW + l
  int b  = bl >> 12;
  int l  = bl & (HW_ - 1);
  int c  = threadIdx.x;                 // 0..95
  float val = in[((long)(b * C_ + c)) * HW_ + l];
  s[c] = val; if (c < 32) s[96 + c] = 0.f;
  __syncthreads();
  for (int off = 64; off > 0; off >>= 1) { if (c < off) s[c] += s[c + off]; __syncthreads(); }
  float mu = s[0] / (float)C_;
  __syncthreads();
  float d = val - mu;
  s[c] = d * d; if (c < 32) s[96 + c] = 0.f;
  __syncthreads();
  for (int off = 64; off > 0; off >>= 1) { if (c < off) s[c] += s[c + off]; __syncthreads(); }
  float var = s[0] / (float)C_;
  out[(long)bl * C_ + c] = d * rsqrtf(var + 1e-5f) * g[c] + bb[c];
}

// LayerNorm over DI=192 of (B*L, 192), in-place, then * silu(z) from proj
__global__ void ln_silu_mul(float* __restrict__ yy, const float* __restrict__ proj,
                            const float* __restrict__ g, const float* __restrict__ bb) {
  __shared__ float s[256];
  int bl = blockIdx.x;
  int d  = threadIdx.x;                 // 0..191
  float val = yy[(long)bl * DI_ + d];
  s[d] = val; if (d < 64) s[192 + d] = 0.f;
  __syncthreads();
  for (int off = 128; off > 0; off >>= 1) { if (d < off) s[d] += s[d + off]; __syncthreads(); }
  float mu = s[0] / (float)DI_;
  __syncthreads();
  float dv = val - mu;
  s[d] = dv * dv; if (d < 64) s[192 + d] = 0.f;
  __syncthreads();
  for (int off = 128; off > 0; off >>= 1) { if (d < off) s[d] += s[d + off]; __syncthreads(); }
  float var = s[0] / (float)DI_;
  float ln = dv * rsqrtf(var + 1e-5f) * g[d] + bb[d];
  float z  = proj[(long)bl * (2 * DI_) + DI_ + d];
  yy[(long)bl * DI_ + d] = ln * (z / (1.f + __expf(-z)));
}

// ---------------------------------------------------------------------------
// Depthwise 3x3 SAME conv.  inMode 0: NCHW, 1: row-strided (b*HW+l)*inStride + c
// act 1 = SiLU
// ---------------------------------------------------------------------------
__global__ void dwconv3x3_k(const float* __restrict__ in, const float* __restrict__ w,
                            const float* __restrict__ bias, float* __restrict__ out,
                            int Ctot, int inMode, int inStride, int act) {
  int idx = blockIdx.x * blockDim.x + threadIdx.x;
  int total = B_ * Ctot * HW_;
  if (idx >= total) return;
  int l  = idx & (HW_ - 1);
  int bc = idx >> 12;
  int c  = bc % Ctot;
  int b  = bc / Ctot;
  int y = l >> 6, x = l & 63;
  float acc = bias ? bias[c] : 0.f;
#pragma unroll
  for (int t = 0; t < 9; ++t) {
    int yy = y + t / 3 - 1, xx = x + t % 3 - 1;
    if ((unsigned)yy >= (unsigned)H_ || (unsigned)xx >= (unsigned)W_) continue;
    int ll = yy * W_ + xx;
    float v = (inMode == 0) ? in[((long)(b * Ctot + c)) * HW_ + ll]
                            : in[((long)(b * HW_ + ll)) * inStride + c];
    acc += v * w[c * 9 + t];
  }
  if (act == 1) acc = acc / (1.f + __expf(-acc));
  out[((long)(b * Ctot + c)) * HW_ + l] = acc;
}

// ---------------------------------------------------------------------------
// Selective scan: 1 thread per (b,k,d); fuses dt projection + softplus + A/D.
// xdbl rows per (b,k): 0..5 dt_raw, 6..21 Bs, 22..37 Cs
// ---------------------------------------------------------------------------
__global__ void scan_k(const float* __restrict__ xc, const float* __restrict__ xdbl,
                       const float* __restrict__ dtw, const float* __restrict__ dtb,
                       const float* __restrict__ Alog, const float* __restrict__ Ds,
                       float* __restrict__ yout) {
  int idx = blockIdx.x * blockDim.x + threadIdx.x;
  if (idx >= B_ * K_ * DI_) return;
  int d = idx % DI_;
  int t = idx / DI_;
  int k = t % K_;
  int b = t / K_;
  float Arow[N_];
#pragma unroll
  for (int n = 0; n < N_; ++n) Arow[n] = -__expf(Alog[(long)(k * DI_ + d) * N_ + n]);
  float dw[R_];
#pragma unroll
  for (int r = 0; r < R_; ++r) dw[r] = dtw[(long)(k * DI_ + d) * R_ + r];
  const float db = dtb[k * DI_ + d];
  const float Dv = Ds[k * DI_ + d];
  const float* xdb = xdbl + ((long)(b * K_ + k)) * 38 * HW_;
  const float* xcb = xc + (long)b * DI_ * HW_ + (long)d * HW_;
  float* yo = yout + ((long)((b * K_ + k) * DI_ + d)) * HW_;
  float h[N_];
#pragma unroll
  for (int n = 0; n < N_; ++n) h[n] = 0.f;
  for (int l = 0; l < HW_; ++l) {
    if (l + 8 < HW_) __builtin_prefetch(xdb + l + 8, 0, 1);
    float u = xcb[scan_gather_idx(k, l)];
    float draw = db;
#pragma unroll
    for (int r = 0; r < R_; ++r) draw += dw[r] * xdb[(long)r * HW_ + l];
    float dt = (draw > 20.f) ? draw : log1pf(__expf(draw));
    float du = dt * u;
    float acc = 0.f;
#pragma unroll
    for (int n = 0; n < N_; ++n) {
      float Bv = xdb[(long)(6 + n) * HW_ + l];
      float Cv = xdb[(long)(22 + n) * HW_ + l];
      h[n] = h[n] * __expf(dt * Arow[n]) + du * Bv;
      acc += h[n] * Cv;
    }
    yo[l] = acc + Dv * u;
  }
}

// Recombine the 4 scan directions -> yy (B*L, DI) row-major
__global__ void yy_combine(const float* __restrict__ y, float* __restrict__ yy) {
  int idx = blockIdx.x * blockDim.x + threadIdx.x;
  int total = B_ * HW_ * DI_;
  if (idx >= total) return;
  int d = idx % DI_;
  int t = idx / DI_;
  int l = t & (HW_ - 1);
  int b = t >> 12;
  int yi = l >> 6, xi = l & 63;
  int lwh = xi * H_ + yi;
  const float* yb = y + (long)b * K_ * DI_ * HW_;
  long cd = (long)d * HW_;
  float v = yb[cd + l] + yb[(long)(2 * DI_) * HW_ + cd + (HW_ - 1 - l)] +
            yb[(long)(1 * DI_) * HW_ + cd + lwh] +
            yb[(long)(3 * DI_) * HW_ + cd + (HW_ - 1 - lwh)];
  yy[(long)t * DI_ + d] = v;
}

// m (NCHW) = mam (NCHW) + part (B*L, C) transposed
__global__ void add_transpose_k(const float* __restrict__ base, const float* __restrict__ part,
                                float* __restrict__ out) {
  int idx = blockIdx.x * blockDim.x + threadIdx.x;
  if (idx >= B_ * C_ * HW_) return;
  int l = idx & (HW_ - 1);
  int bc = idx >> 12;
  int c = bc % C_;
  int b = bc / C_;
  out[idx] = base[idx] + part[((long)(b * HW_ + l)) * C_ + c];
}

// mean over HW per (b,c)
__global__ void mean_hw_k(const float* __restrict__ in, float* __restrict__ out) {
  __shared__ float s[256];
  int tid = threadIdx.x;
  long base = (long)blockIdx.x * HW_;
  float acc = 0.f;
  for (int i = tid; i < HW_; i += 256) acc += in[base + i];
  s[tid] = acc;
  __syncthreads();
  for (int off = 128; off > 0; off >>= 1) { if (tid < off) s[tid] += s[tid + off]; __syncthreads(); }
  if (tid == 0) out[blockIdx.x] = s[0] / (float)HW_;
}

// channel-attention MLP: gate = sigmoid(w2 @ relu(w1 @ gmean + b1) + b2)
__global__ void ca_mlp_k(const float* __restrict__ gmean, const float* __restrict__ w1,
                         const float* __restrict__ b1, const float* __restrict__ w2,
                         const float* __restrict__ b2, float* __restrict__ gate) {
  __shared__ float hid[6];
  int b = blockIdx.x, tid = threadIdx.x;  // 96 threads
  if (tid < 6) {
    float a = b1[tid];
    for (int c = 0; c < C_; ++c) a += w1[tid * C_ + c] * gmean[b * C_ + c];
    hid[tid] = fmaxf(a, 0.f);
  }
  __syncthreads();
  float a = b2[tid];
#pragma unroll
  for (int j = 0; j < 6; ++j) a += w2[tid * 6 + j] * hid[j];
  gate[b * C_ + tid] = 1.f / (1.f + __expf(-a));
}

// dst = src + r * gate[b*C + c]
__global__ void ca_apply_k(const float* __restrict__ src, const float* __restrict__ r,
                           const float* __restrict__ gate, float* __restrict__ dst) {
  int idx = blockIdx.x * blockDim.x + threadIdx.x;
  if (idx >= B_ * C_ * HW_) return;
  int bc = idx >> 12;
  dst[idx] = src[idx] + r[idx] * gate[bc];
}

// l2-normalize rows of length HW, in place; row c of batch b at (b*chanStride+c)*HW
__global__ void l2norm_rows_k(float* __restrict__ x, int chanStride) {
  __shared__ float s[256];
  int tid = threadIdx.x;
  int b = blockIdx.x / C_;
  int c = blockIdx.x % C_;
  float* row = x + ((long)(b * chanStride + c)) * HW_;
  float acc = 0.f;
  for (int i = tid; i < HW_; i += 256) { float v = row[i]; acc += v * v; }
  s[tid] = acc;
  __syncthreads();
  for (int off = 128; off > 0; off >>= 1) { if (tid < off) s[tid] += s[tid + off]; __syncthreads(); }
  float inv = 1.f / fmaxf(sqrtf(s[0]), 1e-12f);
  for (int i = tid; i < HW_; i += 256) row[i] *= inv;
}

// gram: logits[b,h,i,j] = temp[h] * sum_l q[b,h*CH+i,l]*k[b,h*CH+j,l]
__global__ void gram_k(const float* __restrict__ qn, const float* __restrict__ kn,
                       const float* __restrict__ temp, float* __restrict__ logits,
                       int kChanStride) {
  __shared__ float s[256];
  int id = blockIdx.x;
  int j = id % CH_; id /= CH_;
  int i = id % CH_; id /= CH_;
  int h = id % HD_;
  int b = id / HD_;
  const float* qr = qn + ((long)(b * C_ + h * CH_ + i)) * HW_;
  const float* kr = kn + ((long)(b * kChanStride + h * CH_ + j)) * HW_;
  int tid = threadIdx.x;
  float acc = 0.f;
  for (int l = tid; l < HW_; l += 256) acc += qr[l] * kr[l];
  s[tid] = acc;
  __syncthreads();
  for (int off = 128; off > 0; off >>= 1) { if (tid < off) s[tid] += s[tid + off]; __syncthreads(); }
  if (tid == 0) logits[(long)blockIdx.x] = s[0] * temp[h];
}

__global__ void softmax12_k(const float* __restrict__ logits, float* __restrict__ w) {
  int idx = threadIdx.x;             // one thread per (b,h,i), 192 total
  if (idx >= B_ * HD_ * CH_) return;
  const float* row = logits + (long)idx * CH_;
  float mx = row[0];
#pragma unroll
  for (int j = 1; j < CH_; ++j) mx = fmaxf(mx, row[j]);
  float sum = 0.f;
  float e[CH_];
#pragma unroll
  for (int j = 0; j < CH_; ++j) { e[j] = __expf(row[j] - mx); sum += e[j]; }
  float inv = 1.f / sum;
#pragma unroll
  for (int j = 0; j < CH_; ++j) w[(long)idx * CH_ + j] = e[j] * inv;
}

// out[b, h*CH+i, l] = sum_j w[b,h,i,j] * v[b, vOff + h*CH+j, l]
__global__ void attn_apply_k(const float* __restrict__ w, const float* __restrict__ v,
                             float* __restrict__ out, int vChanStride, int vOff) {
  int idx = blockIdx.x * blockDim.x + threadIdx.x;
  if (idx >= B_ * C_ * HW_) return;
  int l = idx & (HW_ - 1);
  int bc = idx >> 12;
  int c = bc % C_;
  int b = bc / C_;
  int h = c / CH_, i = c % CH_;
  const float* wr = w + ((long)((b * HD_ + h) * CH_ + i)) * CH_;
  const float* vb = v + ((long)(b * vChanStride + vOff + h * CH_)) * HW_;
  float acc = 0.f;
#pragma unroll
  for (int j = 0; j < CH_; ++j) acc += wr[j] * vb[(long)j * HW_ + l];
  out[idx] = acc;
}

// concat channels: out (B,2C,HW)
__global__ void concat2_k(const float* __restrict__ a, const float* __restrict__ b2,
                          float* __restrict__ out) {
  int idx = blockIdx.x * blockDim.x + threadIdx.x;
  if (idx >= B_ * 2 * C_ * HW_) return;
  int l = idx & (HW_ - 1);
  int bc = idx >> 12;
  int c = bc % (2 * C_);
  int b = bc / (2 * C_);
  out[idx] = (c < C_) ? a[((long)(b * C_ + c)) * HW_ + l]
                      : b2[((long)(b * C_ + c - C_)) * HW_ + l];
}

// final: sigmoid gate + nan_to_num, writes both tuple outputs concatenated
__global__ void final_gate_k(const float* __restrict__ f, const float* __restrict__ m3,
                             const float* __restrict__ c3, float* __restrict__ out) {
  int idx = blockIdx.x * blockDim.x + threadIdx.x;
  if (idx >= B_ * 2 * C_ * HW_) return;
  int l = idx & (HW_ - 1);
  int bc = idx >> 12;
  int c = bc % (2 * C_);
  int b = bc / (2 * C_);
  float a = 1.f / (1.f + __expf(-f[idx]));
  float v; long o;
  if (c < C_) { v = m3[((long)(b * C_ + c)) * HW_ + l] * a;
                o = ((long)(b * C_ + c)) * HW_ + l; }
  else        { v = a * c3[((long)(b * C_ + c - C_)) * HW_ + l];
                o = (long)B_ * C_ * HW_ + ((long)(b * C_ + c - C_)) * HW_ + l; }
  if (!isfinite(v)) v = 1e-5f;
  out[o] = v;
}

// ---------------------------------------------------------------------------
// Host side
// ---------------------------------------------------------------------------
static inline int cdiv(int a, int b) { return (a + b - 1) / b; }

static void gemm(hipStream_t s, const float* A, const float* Bp, const float* bias,
                 const float* res, float* out, int M, int N, int K,
                 int amode, int lda, int Cin, int bmode, int ldb, int dir,
                 int act, int outmode, int z = 1, long sAz = 0, long sBz = 0, long sOz = 0) {
  dim3 grid(cdiv(M, 64), cdiv(N, 32), z);
  wmma_gemm<<<grid, 128, 0, s>>>(A, Bp, bias, res, out, M, N, K, amode, lda, Cin,
                                 bmode, ldb, dir, act, outmode, sAz, sBz, sOz);
}

extern "C" void kernel_launch(void* const* d_in, const int* in_sizes, int n_in,
                              void* d_out, int out_size, void* d_ws, size_t ws_size,
                              hipStream_t stream) {
  const float* mam       = (const float*)d_in[0];
  const float* cnn       = (const float*)d_in[1];
  const float* ln_g      = (const float*)d_in[2];
  const float* ln_b      = (const float*)d_in[3];
  const float* in_proj_w = (const float*)d_in[4];
  const float* conv_w    = (const float*)d_in[5];
  const float* conv_b    = (const float*)d_in[6];
  const float* x_proj_w  = (const float*)d_in[7];
  const float* dt_w      = (const float*)d_in[8];
  const float* dt_b      = (const float*)d_in[9];
  const float* A_logs    = (const float*)d_in[10];
  const float* Ds        = (const float*)d_in[11];
  const float* out_ln_g  = (const float*)d_in[12];
  const float* out_ln_b  = (const float*)d_in[13];
  const float* out_proj_w= (const float*)d_in[14];
  const float* rcab_w    = (const float*)d_in[15];
  const float* rcab_b    = (const float*)d_in[16];
  const float* ca_w1     = (const float*)d_in[17];
  const float* ca_b1     = (const float*)d_in[18];
  const float* ca_w2     = (const float*)d_in[19];
  const float* ca_b2     = (const float*)d_in[20];
  const float* rg_tail_w = (const float*)d_in[21];
  const float* rg_tail_b = (const float*)d_in[22];
  const float* fus_mam_w = (const float*)d_in[23];
  const float* fus_mam_b = (const float*)d_in[24];
  const float* fus_cnn_w = (const float*)d_in[25];
  const float* fus_cnn_b = (const float*)d_in[26];
  const float* att_temp  = (const float*)d_in[27];
  const float* att_kv_w  = (const float*)d_in[28];
  const float* att_kv_dw = (const float*)d_in[29];
  const float* att_q_w   = (const float*)d_in[30];
  const float* att_q_dw  = (const float*)d_in[31];
  const float* att_proj_w= (const float*)d_in[32];
  const float* fuse_w1   = (const float*)d_in[33];
  const float* fuse_b1   = (const float*)d_in[34];
  const float* fuse_w2   = (const float*)d_in[35];
  const float* fuse_b2   = (const float*)d_in[36];
  float* out = (float*)d_out;
  float* ws  = (float*)d_ws;

  const long IMG  = (long)B_ * C_ * HW_;        // 786432
  const long IMG2 = (long)B_ * 2 * C_ * HW_;    // 1572864

  // ---- Phase A (mamba) arena --------------------------------------------
  float* proj  = ws;                           // (B*L, 384)
  float* hwc   = proj + (long)B_ * HW_ * 2 * DI_;   // (B*L, 96)
  float* xc    = hwc + IMG;                    // (B, DI, HW)
  float* xdbl  = xc + (long)B_ * DI_ * HW_;    // (B, K, 38, HW)
  float* ybuf  = xdbl + (long)B_ * K_ * 38 * HW_;   // (B, K, DI, HW)
  float* yybuf = ybuf + (long)B_ * K_ * DI_ * HW_;  // (B*L, DI)
  float* mpart = yybuf + (long)B_ * HW_ * DI_; // (B*L, C)
  // ---- Phase B (cnn/attn) arena, reuses phase-A space from ws ------------
  float* cb    = ws;
  float* r1    = cb + IMG;
  float* r2    = r1 + IMG;
  float* cfeat = r2 + IMG;
  float* m2    = cfeat + IMG;
  float* c2    = m2 + IMG;
  float* kvb   = c2 + IMG;        // (B, 2C, HW)
  float* kvd   = kvb + IMG2;
  float* q1    = kvd + IMG2;
  float* qd    = q1 + IMG;
  float* attno = qd + IMG;
  float* m3    = attno + IMG;
  float* c3    = m3 + IMG;
  float* catb  = c3 + IMG;        // (B, 2C, HW)
  float* f1    = catb + IMG2;
  float* f2    = f1 + IMG;        // (B, 2C, HW)
  // persistent across both phases:
  float* mbuf  = f2 + IMG2;       // m (B,C,HW) -- past both arenas' peaks
  float* gmean = mbuf + IMG;      // B*C
  float* gate  = gmean + 256;
  float* logit = gate + 256;      // 2304
  float* attw  = logit + 4096;    // 2304
  (void)n_in; (void)in_sizes; (void)out_size; (void)ws_size; (void)mpart;

  const int EW = 256;             // elementwise block

  // ======================= Mamba branch ===================================
  ln_nchw_to_nlc<<<B_ * HW_, 96, 0, stream>>>(mam, ln_g, ln_b, hwc);
  // in_proj: (B*L,96) @ (384,96)^T -> proj (B*L,384)
  gemm(stream, hwc, in_proj_w, nullptr, nullptr, proj,
       B_ * HW_, 2 * DI_, C_, 0, C_, 0, 0, 0, 0, 0, 0);
  // depthwise conv + SiLU on xi (first DI cols of proj) -> xc NCHW
  dwconv3x3_k<<<cdiv(B_ * DI_ * HW_, EW), EW, 0, stream>>>(
      proj, conv_w, conv_b, xc, DI_, 1, 2 * DI_, 1);
  // x_dbl[b,k] = x_proj_w[k] (38x192) @ xs[b,k] (192xHW), xs gathered from xc
  for (int k = 0; k < K_; ++k) {
    gemm(stream, x_proj_w + (long)k * 38 * DI_, xc, nullptr, nullptr,
         xdbl + (long)k * 38 * HW_, 38, HW_, DI_,
         0, DI_, 0, 2, 0, k, 0, 0,
         B_, 0, (long)DI_ * HW_, (long)K_ * 38 * HW_);
  }
  // selective scan (fused dt-proj + softplus + A/D)
  scan_k<<<cdiv(B_ * K_ * DI_, 128), 128, 0, stream>>>(
      xc, xdbl, dt_w, dt_b, A_logs, Ds, ybuf);
  yy_combine<<<cdiv((int)(B_ * HW_ * DI_), EW), EW, 0, stream>>>(ybuf, yybuf);
  ln_silu_mul<<<B_ * HW_, 192, 0, stream>>>(yybuf, proj, out_ln_g, out_ln_b);
  // out_proj: (B*L,192) @ (96,192)^T -> mpart
  gemm(stream, yybuf, out_proj_w, nullptr, nullptr, mpart,
       B_ * HW_, C_, DI_, 0, DI_, 0, 0, 0, 0, 0, 0);
  add_transpose_k<<<cdiv((int)IMG, EW), EW, 0, stream>>>(mam, mpart, mbuf);

  // ======================= CNN branch (RCAB x2) ===========================
  const float* src = cnn;
  for (int i = 0; i < 2; ++i) {
    const float* w0 = rcab_w + (long)(i * 2 + 0) * C_ * C_ * 9;
    const float* w1 = rcab_w + (long)(i * 2 + 1) * C_ * C_ * 9;
    const float* b0 = rcab_b + (i * 2 + 0) * C_;
    const float* b1 = rcab_b + (i * 2 + 1) * C_;
    gemm(stream, src, w0, b0, nullptr, r1, B_ * HW_, C_, C_ * 9, 1, 0, C_, 0, 0, 0, 1, 1);
    gemm(stream, r1, w1, b1, nullptr, r2, B_ * HW_, C_, C_ * 9, 1, 0, C_, 0, 0, 0, 0, 1);
    mean_hw_k<<<B_ * C_, 256, 0, stream>>>(r2, gmean);
    ca_mlp_k<<<B_, 96, 0, stream>>>(gmean, ca_w1 + (long)i * 6 * C_, ca_b1 + i * 6,
                                    ca_w2 + (long)i * C_ * 6, ca_b2 + i * C_, gate);
    ca_apply_k<<<cdiv((int)IMG, EW), EW, 0, stream>>>(src, r2, gate, cb);
    src = cb;
  }
  // cfeat = cnn + conv3x3(cb, rg_tail)
  gemm(stream, cb, rg_tail_w, rg_tail_b, cnn, cfeat, B_ * HW_, C_, C_ * 9, 1, 0, C_, 0, 0, 0, 0, 1);

  // ======================= Fusion =========================================
  gemm(stream, mbuf,  fus_mam_w, fus_mam_b, nullptr, m2, B_ * HW_, C_, C_ * 9, 1, 0, C_, 0, 0, 0, 0, 1);
  gemm(stream, cfeat, fus_cnn_w, fus_cnn_b, nullptr, c2, B_ * HW_, C_, C_ * 9, 1, 0, C_, 0, 0, 0, 0, 1);

  // attention(x, y) called twice
  for (int call = 0; call < 2; ++call) {
    const float* x = (call == 0) ? m2 : c2;
    const float* y = (call == 0) ? c2 : m3;
    float* dst     = (call == 0) ? m3 : c3;
    // kv = 1x1(y) -> (B,2C,HW); then depthwise 3x3
    gemm(stream, y, att_kv_w + (long)call * 2 * C_ * C_, nullptr, nullptr, kvb,
         B_ * HW_, 2 * C_, C_, 2, 0, C_, 0, 0, 0, 0, 1);
    dwconv3x3_k<<<cdiv((int)IMG2, EW), EW, 0, stream>>>(
        kvb, att_kv_dw + (long)call * 2 * C_ * 9, nullptr, kvd, 2 * C_, 0, 0, 0);
    // q = 1x1(x) -> dw 3x3
    gemm(stream, x, att_q_w + (long)call * C_ * C_, nullptr, nullptr, q1,
         B_ * HW_, C_, C_, 2, 0, C_, 0, 0, 0, 0, 1);
    dwconv3x3_k<<<cdiv((int)IMG, EW), EW, 0, stream>>>(
        q1, att_q_dw + (long)call * C_ * 9, nullptr, qd, C_, 0, 0, 0);
    // l2norm q rows and k rows (k = first C channels of kvd)
    l2norm_rows_k<<<B_ * C_, 256, 0, stream>>>(qd, C_);
    l2norm_rows_k<<<B_ * C_, 256, 0, stream>>>(kvd, 2 * C_);
    gram_k<<<B_ * HD_ * CH_ * CH_, 256, 0, stream>>>(qd, kvd, att_temp + call * HD_,
                                                     logit, 2 * C_);
    softmax12_k<<<1, 192, 0, stream>>>(logit, attw);
    attn_apply_k<<<cdiv((int)IMG, EW), EW, 0, stream>>>(attw, kvd, attno, 2 * C_, C_);
    // proj 1x1 + residual x
    gemm(stream, attno, att_proj_w + (long)call * C_ * C_, nullptr, x, dst,
         B_ * HW_, C_, C_, 2, 0, C_, 0, 0, 0, 0, 1);
  }

  concat2_k<<<cdiv((int)IMG2, EW), EW, 0, stream>>>(m3, c3, catb);
  gemm(stream, catb, fuse_w1, fuse_b1, nullptr, f1, B_ * HW_, C_, 2 * C_ * 9, 1, 0, 2 * C_, 0, 0, 0, 0, 1);
  gemm(stream, f1, fuse_w2, fuse_b2, nullptr, f2, B_ * HW_, 2 * C_, C_ * 9, 1, 0, C_, 0, 0, 0, 0, 1);
  final_gate_k<<<cdiv((int)IMG2, EW), EW, 0, stream>>>(f2, m3, c3, out);
}
